// Attention_80908593923409
// MI455X (gfx1250) — matrix operand order repaired and checked by reading, not compiled, not run on hardware
//
#include <hip/hip_runtime.h>
#include <math.h>

// ---------------- types ----------------
typedef __attribute__((ext_vector_type(8)))  float  v8f;
typedef __attribute__((ext_vector_type(16))) __bf16 v16bf;
typedef __attribute__((ext_vector_type(8)))  __bf16 v8bf;

// ---------------- problem constants (from reference setup_inputs) ----------------
constexpr int BB   = 8;     // batch
constexpr int TT   = 16;    // new tokens
constexpr int DD   = 1536;  // model dim
constexpr int NH   = 12;    // q heads
constexpr int KVH  = 2;     // kv heads
constexpr int GG   = 6;     // NH / KVH
constexpr int HD   = 128;   // head dim
constexpr int SS   = 8192;  // cache length
constexpr int SPLITS = 16;  // flash-decoding splits of S
constexpr int SCHUNK = SS / SPLITS; // 512
constexpr int LROW = 132;   // padded LDS row (floats): b128 reads start 4 banks apart
constexpr int NPART = BB * KVH * GG * SPLITS; // 1536 partial blocks

constexpr float SCALE    = 0.08838834764831845f; // H^-0.5
constexpr float MASKVAL  = -3.0e38f;             // ~bf16 min
constexpr float LN_THETA = 13.815510557964274f;  // ln(1e6)

// ---------------- workspace layout (bytes) ----------------
constexpr size_t OFF_QF    = 0;                         // f32 (b,t,N*H)   128x1536
constexpr size_t OFF_KF    = OFF_QF    + 128*1536*4;    // f32 (b,t,K*H)   128x256
constexpr size_t OFF_VF    = OFF_KF    + 128*256*4;     // f32 (b,t,K*H)
constexpr size_t OFF_QBF   = OFF_VF    + 128*256*4;     // bf16 (b,t,N,H) rotated
constexpr size_t OFF_KBF   = OFF_QBF   + 128*1536*2;    // bf16 (b,t,K,H) rotated
constexpr size_t OFF_VBF   = OFF_KBF   + 128*256*2;     // bf16 (b,t,K,H)
constexpr size_t OFF_OBF   = OFF_VBF   + 128*256*2;     // bf16 (b,t,N,H) attn out
constexpr size_t OFF_PM    = OFF_OBF   + 128*1536*2;    // f32 partial max  [NPART*16]
constexpr size_t OFF_PL    = OFF_PM    + (size_t)NPART*16*4;
constexpr size_t OFF_PACC  = OFF_PL    + (size_t)NPART*16*4; // f32 [NPART*16*128]

// ---------------- WMMA helpers ----------------
__device__ inline v8f wmma_bf16(v16bf a, v16bf b, v8f c) {
  return __builtin_amdgcn_wmma_f32_16x16x32_bf16(false, a, false, b, (short)0, c,
                                                 false, false);
}

// A fragment (16x32 bf16) from a row-major bf16 row pointer (points at row, k0).
// Per-lane K run: [half*16, half*16+16).
__device__ inline v16bf load_a_bf16(const __bf16* rowk, int half) {
  v8bf lo = *(const v8bf*)(rowk + half * 16);
  v8bf hi = *(const v8bf*)(rowk + half * 16 + 8);
  return __builtin_shufflevector(lo, hi, 0,1,2,3,4,5,6,7,8,9,10,11,12,13,14,15);
}

// 16 strided fp32 -> bf16 vector (B fragments; stride 1 => contiguous loads)
__device__ inline v16bf cvt16(const float* base, int stride) {
  v16bf f;
#pragma unroll
  for (int j = 0; j < 16; ++j) f[j] = (__bf16)base[j * stride];
  return f;
}

// ---------------- DPP16 butterfly reductions within 16-lane rows ----------------
// quad_perm[1,0,3,2]=0xB1, quad_perm[2,3,0,1]=0x4E, row_half_mirror=0x141, row_mirror=0x140
__device__ inline float dpp_max16(float v) {
  v = fmaxf(v, __int_as_float(__builtin_amdgcn_update_dpp(0, __float_as_int(v), 0xB1, 0xF, 0xF, true)));
  v = fmaxf(v, __int_as_float(__builtin_amdgcn_update_dpp(0, __float_as_int(v), 0x4E, 0xF, 0xF, true)));
  v = fmaxf(v, __int_as_float(__builtin_amdgcn_update_dpp(0, __float_as_int(v), 0x141, 0xF, 0xF, true)));
  v = fmaxf(v, __int_as_float(__builtin_amdgcn_update_dpp(0, __float_as_int(v), 0x140, 0xF, 0xF, true)));
  return v;
}
__device__ inline float dpp_sum16(float v) {
  v += __int_as_float(__builtin_amdgcn_update_dpp(0, __float_as_int(v), 0xB1, 0xF, 0xF, true));
  v += __int_as_float(__builtin_amdgcn_update_dpp(0, __float_as_int(v), 0x4E, 0xF, 0xF, true));
  v += __int_as_float(__builtin_amdgcn_update_dpp(0, __float_as_int(v), 0x141, 0xF, 0xF, true));
  v += __int_as_float(__builtin_amdgcn_update_dpp(0, __float_as_int(v), 0x140, 0xF, 0xF, true));
  return v;
}

// ---------------- WMMA GEMM: full 128x16 column strip per wave ----------------
// B fragment loaded once per k-step and reused across all 8 row tiles
// (weights are streamed exactly once). ABF=1: A bf16; ABF=0: A fp32 (cvt on load).
template <int ABF>
__global__ void gemm_wmma(const float* __restrict__ Af, const __bf16* __restrict__ Ab,
                          int lda, const float* __restrict__ B, int ldb,
                          const float* __restrict__ bias, float* __restrict__ C,
                          int ldc, int Kdim) {
  const int lane = threadIdx.x & 31;
  const int half = lane >> 4, l16 = lane & 15;
  const int bcol = blockIdx.x * 16 + l16;

  v8f acc[8];
#pragma unroll
  for (int rt = 0; rt < 8; ++rt) acc[rt] = (v8f){};

  for (int k0 = 0; k0 < Kdim; k0 += 32) {
    const v16bf bfrag = cvt16(B + (k0 + 16 * half) * ldb + bcol, ldb);
#pragma unroll
    for (int rt = 0; rt < 8; ++rt) {
      const int arow = rt * 16 + l16;
      v16bf a;
      if (ABF) {
        a = load_a_bf16(Ab + arow * lda + k0, half);
      } else {
        const float* ap = Af + arow * lda + k0;
        v16bf t;
#pragma unroll
        for (int j = 0; j < 8; ++j) t[j] = (__bf16)ap[half * 16 + j];
#pragma unroll
        for (int j = 0; j < 8; ++j) t[8 + j] = (__bf16)ap[half * 16 + 8 + j];
        a = t;
      }
      acc[rt] = wmma_bf16(a, bfrag, acc[rt]);
    }
  }
  const float bv = bias ? bias[bcol] : 0.0f;
#pragma unroll
  for (int rt = 0; rt < 8; ++rt)
#pragma unroll
    for (int r = 0; r < 8; ++r)
      C[(rt * 16 + r + 8 * half) * ldc + bcol] = acc[rt][r] + bv;
}

// ---------------- RoPE + bf16 pack ----------------
__global__ void rope_kernel(const float* __restrict__ qf, const float* __restrict__ kf,
                            const float* __restrict__ vf, const int* __restrict__ seg,
                            const int* __restrict__ curp, __bf16* __restrict__ qbf,
                            __bf16* __restrict__ kbf, __bf16* __restrict__ vbf) {
  const int bt = blockIdx.x;           // (b*16 + t)
  const int b = bt >> 4, t = bt & 15;
  const int cur = *curp;
  int cnt = 0;
  for (int i = 0; i <= t; ++i) cnt += (seg[b * 16 + i] != 0);
  const int pos = max(cnt - 1, 0) + cur;
  const int tid = threadIdx.x;         // 128 threads

  // q: 12 heads * 64 pairs
  for (int idx = tid; idx < NH * 64; idx += 128) {
    const int n = idx >> 6, j = idx & 63;
    const float inv_ts = __expf(-((float)j * (1.0f / 64.0f)) * LN_THETA);
    float s, c;
    __sincosf((float)pos * inv_ts, &s, &c);
    const float* qp = qf + bt * 1536 + n * 128;
    __bf16* qo = qbf + bt * 1536 + n * 128;
    const float x1 = qp[j], x2 = qp[j + 64];
    qo[j]      = (__bf16)(x1 * c - x2 * s);
    qo[j + 64] = (__bf16)(x2 * c + x1 * s);
  }
  // k: 2 heads * 64 pairs
  {
    const int kh = tid >> 6, j = tid & 63;
    const float inv_ts = __expf(-((float)j * (1.0f / 64.0f)) * LN_THETA);
    float s, c;
    __sincosf((float)pos * inv_ts, &s, &c);
    const float* kp = kf + bt * 256 + kh * 128;
    __bf16* ko = kbf + bt * 256 + kh * 128;
    const float x1 = kp[j], x2 = kp[j + 64];
    ko[j]      = (__bf16)(x1 * c - x2 * s);
    ko[j + 64] = (__bf16)(x2 * c + x1 * s);
  }
  // v: straight convert, 256 elems
  for (int idx = tid; idx < 256; idx += 128)
    vbf[bt * 256 + idx] = (__bf16)vf[bt * 256 + idx];
}

// ---------------- flash-decoding attention ----------------
// grid (SPLITS, KVH, BB); 6 waves/block. Double-buffered LDS pipeline: chunk
// i+1 is fetched with global_load_async_to_lds_b128 (ASYNCcnt) while all six
// waves run WMMA on chunk i. Online softmax uses DPP16 row butterflies.
__global__ __launch_bounds__(192) void attn_kernel(
    const float* __restrict__ k_cache, const float* __restrict__ v_cache,
    const __bf16* __restrict__ qbf, const __bf16* __restrict__ kbf,
    const __bf16* __restrict__ vbf, const int* __restrict__ seg,
    const int* __restrict__ start_ind, const int* __restrict__ curp,
    float* __restrict__ pm, float* __restrict__ pl, float* __restrict__ pacc) {
  const int b = blockIdx.z, kh = blockIdx.y, split = blockIdx.x;
  const int wave = threadIdx.x >> 5;   // g
  const int lane = threadIdx.x & 31;
  const int half = lane >> 4, l16 = lane & 15;
  const int n = kh * GG + wave;
  const int cur = *curp;

  int start_b = start_ind[b];
  if (start_b < 0) {
    start_b = 0;
    for (int i = 0; i < TT; ++i)
      if (seg[b * TT + i] != 0) { start_b = i; break; }
  }
  int segrow[8];
#pragma unroll
  for (int r = 0; r < 8; ++r) segrow[r] = seg[b * TT + r + 8 * half];

  // preload Q A-fragments (rows = token index = l16)
  const __bf16* qrow = qbf + ((b * TT + l16) * NH + n) * HD;
  v16bf qfrag[4];
#pragma unroll
  for (int hc = 0; hc < 4; ++hc) qfrag[hc] = load_a_bf16(qrow + hc * 32, half);

  float m_run[8], l_run[8];
  v8f acc[8];
#pragma unroll
  for (int r = 0; r < 8; ++r) { m_run[r] = -1.0e30f; l_run[r] = 0.0f; }
#pragma unroll
  for (int ht = 0; ht < 8; ++ht) acc[ht] = (v8f){};

  __shared__ float  kbuf[2][32 * LROW];       // ping-pong K chunks (fp32)
  __shared__ float  vbuf[2][32 * LROW];       // ping-pong V chunks
  __shared__ __bf16 pbuf[6 * 16 * 32];        // per-wave P tiles
  __bf16* pw = pbuf + wave * (16 * 32);

  const int s_end = cur + TT;                       // keys >= this always masked
  const int lim   = min(split * SCHUNK + SCHUNK, s_end);
  const int s_first = max(split * SCHUNK, (start_b >> 5) << 5);

  // stage one 32-key chunk (K and V) into LDS buffer bufi
  auto stage_chunk = [&](int sc, int bufi) {
#pragma unroll
    for (int isv = 0; isv < 2; ++isv) {
      const float*  gbase  = (isv ? v_cache : k_cache) + (b * SS * KVH + kh) * HD;
      const __bf16* wsbase = (isv ? vbf : kbf);
      float*        lbase  = isv ? vbuf[bufi] : kbuf[bufi];
      for (int idx = threadIdx.x; idx < 1024; idx += 192) {
        const int r = idx >> 5, h4 = idx & 31;      // row 0..31, float4 0..31
        const int s_glob = sc + r;
        float* dst = lbase + r * LROW + h4 * 4;
        if (s_glob < cur) {
          const unsigned ldsoff = (unsigned)(unsigned long long)(void*)dst;
          const unsigned voff = (unsigned)(s_glob * (KVH * HD * 4) + h4 * 16);
          asm volatile("global_load_async_to_lds_b128 %0, %1, %2 offset:0"
                       :: "v"(ldsoff), "v"(voff), "s"(gbase) : "memory");
        } else {
          const int srel = min(s_glob - cur, TT - 1);
          const __bf16* src = wsbase + ((b * TT + srel) * KVH + kh) * HD + h4 * 4;
          float4 t4;
          t4.x = (float)src[0]; t4.y = (float)src[1];
          t4.z = (float)src[2]; t4.w = (float)src[3];
          *(float4*)dst = t4;
        }
      }
    }
  };

  if (s_first < lim) stage_chunk(s_first, 0);

  int buf = 0;
  for (int s0 = s_first; s0 < lim; s0 += 32, buf ^= 1) {
    // previous issues (this buffer) must have landed; then prefetch next chunk
    asm volatile("s_wait_asynccnt 0x0" ::: "memory");
    __syncthreads();
    if (s0 + 32 < lim) stage_chunk(s0 + 32, buf ^ 1);

    const float* kb = kbuf[buf];
    const float* vb = vbuf[buf];

    // ---- logits: two 16x16 tiles via WMMA over 4 h-chunks ----
    float tv[2][8];
#pragma unroll
    for (int tile = 0; tile < 2; ++tile) {
      const int sb = s0 + tile * 16;
      v8f lg = {};
      if (sb < s_end) {
        const int sloc = tile * 16 + l16;
#pragma unroll
        for (int hc = 0; hc < 4; ++hc) {
          const int hb = hc * 32 + 16 * half;
          const v16bf kfrag = cvt16(kb + sloc * LROW + hb, 1);
          lg = wmma_bf16(qfrag[hc], kfrag, lg);
        }
      }
#pragma unroll
      for (int r = 0; r < 8; ++r) {
        const int tq = r + 8 * half;
        const int scol = sb + l16;
        const int kvseg = (scol >= start_b && scol < s_end) ? 1 : 0;
        const bool valid = (sb < s_end) && (scol <= cur + tq) && (kvseg == segrow[r]);
        tv[tile][r] = valid ? lg[r] * SCALE : MASKVAL;
      }
    }

    // ---- online softmax update (DPP16 row butterflies) ----
    float fac[8];
#pragma unroll
    for (int r = 0; r < 8; ++r) {
      const float v = dpp_max16(fmaxf(tv[0][r], tv[1][r]));
      const float m_new = fmaxf(m_run[r], v);
      fac[r] = __expf(m_run[r] - m_new);
      m_run[r] = m_new;
    }
#pragma unroll
    for (int tile = 0; tile < 2; ++tile)
#pragma unroll
      for (int r = 0; r < 8; ++r) {
        const float p = __expf(tv[tile][r] - m_run[r]);
        tv[tile][r] = p;
        pw[(r + 8 * half) * 32 + tile * 16 + l16] = (__bf16)p;
      }
#pragma unroll
    for (int r = 0; r < 8; ++r)
      l_run[r] = l_run[r] * fac[r] + dpp_sum16(tv[0][r] + tv[1][r]);
#pragma unroll
    for (int ht = 0; ht < 8; ++ht)
#pragma unroll
      for (int r = 0; r < 8; ++r) acc[ht][r] *= fac[r];

    asm volatile("s_wait_dscnt 0x0" ::: "memory"); // P stores -> P loads (same wave)

    // ---- O += P(16x32) * V(32x128) ----
    const v16bf pa = load_a_bf16(pw + l16 * 32, half);
#pragma unroll
    for (int ht = 0; ht < 8; ++ht) {
      const int hcol = ht * 16 + l16;
      v16bf vfrag;
#pragma unroll
      for (int j = 0; j < 16; ++j)
        vfrag[j] = (__bf16)vb[(16 * half + j) * LROW + hcol];
      acc[ht] = wmma_bf16(pa, vfrag, acc[ht]);
    }
    __syncthreads();   // all waves done with this buffer before it is restaged
  }

  // ---- write per-split partials ----
  const int p = ((b * KVH + kh) * GG + wave) * SPLITS + split;
  if (l16 == 0) {
#pragma unroll
    for (int r = 0; r < 8; ++r) {
      const int row = r + 8 * half;
      pm[p * 16 + row] = m_run[r];
      pl[p * 16 + row] = l_run[r];
    }
  }
#pragma unroll
  for (int ht = 0; ht < 8; ++ht)
#pragma unroll
    for (int r = 0; r < 8; ++r) {
      const int row = r + 8 * half;
      pacc[(p * 16 + row) * HD + ht * 16 + l16] = acc[ht][r];
    }
}

// ---------------- split combine ----------------
__global__ void combine_kernel(const float* __restrict__ pm, const float* __restrict__ pl,
                               const float* __restrict__ pacc, __bf16* __restrict__ obf) {
  const int row = blockIdx.x;          // bkg*16 + t, bkg = (b*KVH+kh)*GG+g
  const int h = threadIdx.x;           // 128
  const int t = row & 15, bkg = row >> 4;
  const int g = bkg % GG, kh = (bkg / GG) % KVH, b = bkg / (GG * KVH);

  float M = -3.0e38f;
  for (int s = 0; s < SPLITS; ++s)
    M = fmaxf(M, pm[(bkg * SPLITS + s) * 16 + t]);
  float L = 0.0f, o = 0.0f;
  for (int s = 0; s < SPLITS; ++s) {
    const float w = __expf(pm[(bkg * SPLITS + s) * 16 + t] - M);
    L += pl[(bkg * SPLITS + s) * 16 + t] * w;
    o += pacc[((bkg * SPLITS + s) * 16 + t) * HD + h] * w;
  }
  const float out = (L > 1e-30f) ? o / L : 0.0f;
  const int n = kh * GG + g;
  obf[((b * TT + t) * NH + n) * HD + h] = (__bf16)out;
}

// ---------------- launcher ----------------
extern "C" void kernel_launch(void* const* d_in, const int* in_sizes, int n_in,
                              void* d_out, int out_size, void* d_ws, size_t ws_size,
                              hipStream_t stream) {
  const float* x        = (const float*)d_in[0];
  const float* k_cache  = (const float*)d_in[1];
  const float* v_cache  = (const float*)d_in[2];
  const float* wq       = (const float*)d_in[3];
  const float* bq       = (const float*)d_in[4];
  const float* wk       = (const float*)d_in[5];
  const float* bk       = (const float*)d_in[6];
  const float* wv       = (const float*)d_in[7];
  const float* bv       = (const float*)d_in[8];
  const float* wo       = (const float*)d_in[9];
  const int*   seg      = (const int*)d_in[10];
  const int*   startp   = (const int*)d_in[11];
  const int*   curp     = (const int*)d_in[12];
  float*       out      = (float*)d_out;

  char* w = (char*)d_ws;
  float*  qf   = (float*)(w + OFF_QF);
  float*  kf   = (float*)(w + OFF_KF);
  float*  vf   = (float*)(w + OFF_VF);
  __bf16* qbf  = (__bf16*)(w + OFF_QBF);
  __bf16* kbf  = (__bf16*)(w + OFF_KBF);
  __bf16* vbf  = (__bf16*)(w + OFF_VBF);
  __bf16* obf  = (__bf16*)(w + OFF_OBF);
  float*  pm   = (float*)(w + OFF_PM);
  float*  pl   = (float*)(w + OFF_PL);
  float*  pacc = (float*)(w + OFF_PACC);

  // 1-3) QKV projections (M=128 rows = b*t; weights streamed once)
  gemm_wmma<0><<<dim3(NH * HD / 16), 32, 0, stream>>>(
      x, nullptr, DD, wq, NH * HD, bq, qf, NH * HD, DD);
  gemm_wmma<0><<<dim3(KVH * HD / 16), 32, 0, stream>>>(
      x, nullptr, DD, wk, KVH * HD, bk, kf, KVH * HD, DD);
  gemm_wmma<0><<<dim3(KVH * HD / 16), 32, 0, stream>>>(
      x, nullptr, DD, wv, KVH * HD, bv, vf, KVH * HD, DD);

  // 4) RoPE + bf16 pack
  rope_kernel<<<BB * TT, 128, 0, stream>>>(qf, kf, vf, seg, curp, qbf, kbf, vbf);

  // 5) flash-decoding attention (double-buffered async K/V staging)
  attn_kernel<<<dim3(SPLITS, KVH, BB), 192, 0, stream>>>(
      k_cache, v_cache, qbf, kbf, vbf, seg, startp, curp, pm, pl, pacc);

  // 6) combine splits -> bf16 attn output
  combine_kernel<<<BB * KVH * GG * TT, HD, 0, stream>>>(pm, pl, pacc, obf);

  // 7) output projection -> d_out (fp32)
  gemm_wmma<1><<<dim3(DD / 16), 32, 0, stream>>>(
      nullptr, obf, NH * HD, wo, DD, nullptr, out, DD, NH * HD);
}
